// WindowAttention_74887049773760
// MI455X (gfx1250) — compile-verified, matrix operand order checked by reading, not run on hardware
//
#include <hip/hip_runtime.h>
#include <hip/hip_bf16.h>
#include <math.h>

// ---------------------------------------------------------------------------
// Fused window attention for gfx1250 (MI455X), wave32, WMMA bf16 path.
//   B=4096 windows, N=49 (pad 64), C=512, H=16 heads, hd=32, WS=7
// One workgroup (8 wave32) per window; everything after the x load lives in
// LDS (288 KB of the WGP's 320 KB). All matmuls use v_wmma_f32_16x16x32_bf16.
// ---------------------------------------------------------------------------

typedef __bf16 bf16_t;
typedef __attribute__((ext_vector_type(16))) __bf16 v16bf;
typedef __attribute__((ext_vector_type(8)))  __bf16 v8bf;
typedef __attribute__((ext_vector_type(8)))  float  v8f;

#define NTOK   49
#define NP     64
#define CDIM   512
#define HEADS  16
#define HD     32
#define WS7    7
#define WAVE_LDS 28672                      // per-wave block bytes
#define SMEM_BYTES (65536 + 8 * WAVE_LDS)   // 294912 = 288 KB

__device__ __forceinline__ v8f vzero8() {
  v8f v;
#pragma unroll
  for (int i = 0; i < 8; ++i) v[i] = 0.0f;
  return v;
}

// Load one 16x(K=32) bf16 fragment (A-operand layout; B-operand uses the same
// per-lane addressing against a [n][k] / transposed surface).
// Per CDNA5 ISA 7.12.2: lane L<16 holds row M=L, K={0..7,16..23};
// lane L+16 holds row M=L, K={8..15,24..31}. Two 16B chunks per lane.
__device__ __forceinline__ v16bf frag_ld(const bf16_t* __restrict__ base,
                                         int stride, int lane) {
  const int r  = lane & 15;
  const int kb = (lane >> 4) << 3;  // 0 or 8
  const bf16_t* p0 = base + r * stride + kb;
  const bf16_t* p1 = base + r * stride + 16 + kb;
  v8bf lo = *(const v8bf*)p0;
  v8bf hi = *(const v8bf*)p1;
  return __builtin_shufflevector(lo, hi, 0, 1, 2, 3, 4, 5, 6, 7,
                                         8, 9, 10, 11, 12, 13, 14, 15);
}

__device__ __forceinline__ v8f wmma_bf16(v16bf a, v16bf b, v8f c) {
  // (neg_a, A, neg_b, B, c_mod, C, reuse_a, reuse_b)
  return __builtin_amdgcn_wmma_f32_16x16x32_bf16(false, a, false, b,
                                                 (short)0, c, false, false);
}

// ---------------------------------------------------------------------------
// Prep: transpose fp32 weights -> bf16 [out_col][k] so WMMA B-fragments are
// contiguous along K. 2 MB total, stays hot in the 192 MB L2 for all 4096 WGs.
// ---------------------------------------------------------------------------
__global__ void prep_weights(const float* __restrict__ w_qkv,   // [512][1536]
                             const float* __restrict__ w_proj,  // [512][512]
                             bf16_t* __restrict__ wqkvT,        // [1536][512]
                             bf16_t* __restrict__ wprojT) {     // [512][512]
  const int i = blockIdx.x * 256 + threadIdx.x;
  if (i < 3 * CDIM * CDIM) {                 // 1536*512
    const int j = i >> 9, k = i & (CDIM - 1);
    wqkvT[i] = (bf16_t)w_qkv[k * (3 * CDIM) + j];
  }
  if (i < CDIM * CDIM) {
    const int j = i >> 9, k = i & (CDIM - 1);
    wprojT[i] = (bf16_t)w_proj[k * CDIM + j];
  }
}

// ---------------------------------------------------------------------------
// Main fused kernel.
// ---------------------------------------------------------------------------
__global__ __launch_bounds__(256, 1)
void win_attn_fused(const float* __restrict__ x,          // [B][49][512]
                    const bf16_t* __restrict__ wqkvT,     // [1536][512]
                    const float* __restrict__ b_qkv,      // [1536]
                    const bf16_t* __restrict__ wprojT,    // [512][512]
                    const float* __restrict__ b_proj,     // [512]
                    const float* __restrict__ bias_table, // [169][16]
                    float* __restrict__ out) {            // [B][49][512]
  extern __shared__ __align__(16) char smem[];
  bf16_t* sX = (bf16_t*)smem;                       // [64][512] (later out_heads)

  const int tid  = threadIdx.x;
  const int wave = tid >> 5;
  const int lane = tid & 31;
  const int b    = blockIdx.x;

  char*   wb  = smem + 65536 + wave * WAVE_LDS;     // per-wave private block
  bf16_t* sQ  = (bf16_t*)wb;                        // [64][32]
  bf16_t* sK  = (bf16_t*)(wb + 4096);               // [64][32]
  bf16_t* sVT = (bf16_t*)(wb + 8192);               // [32][64] (V transposed)
  float*  sS  = (float*)(wb + 12288);               // [64][64] scores
  bf16_t* sP  = (bf16_t*)wb;                        // [64][64] aliases sQ+sK

  const int nlo = lane & 15;
  const int mhi = (lane >> 4) << 3;

  // ---- Phase A: stage x[b] -> LDS bf16, rows 49..63 zeroed -----------------
  for (int i = tid; i < NP * CDIM; i += 256) {
    const int r = i >> 9, c = i & (CDIM - 1);
    float v = (r < NTOK) ? x[((size_t)b * NTOK + r) * CDIM + c] : 0.0f;
    sX[i] = (bf16_t)v;
  }
  __syncthreads();

  v8f oacc[2][4][2];
  const float qscale = 0.04419417382415922f;        // 512^-0.5 (full C, as ref)

#pragma unroll
  for (int hIter = 0; hIter < 2; ++hIter) {
    const int h = wave + 8 * hIter;

    // ---- Phase B: per-head QKV GEMM [64,512] x [512,32] --------------------
#pragma unroll
    for (int t = 0; t < 3; ++t) {                   // 0=Q 1=K 2=V
      const int cb = t * CDIM + h * HD;             // row base into wqkvT
      v8f acc[4][2];
#pragma unroll
      for (int mt = 0; mt < 4; ++mt) { acc[mt][0] = vzero8(); acc[mt][1] = vzero8(); }

      for (int kk = 0; kk < CDIM / 32; ++kk) {      // rolled K loop
        const int kb = kk * 32;
        v16bf bf0 = frag_ld(wqkvT + (size_t)(cb)      * CDIM + kb, CDIM, lane);
        v16bf bf1 = frag_ld(wqkvT + (size_t)(cb + 16) * CDIM + kb, CDIM, lane);
#pragma unroll
        for (int mt = 0; mt < 4; ++mt) {
          v16bf af = frag_ld(sX + mt * 16 * CDIM + kb, CDIM, lane);
          acc[mt][0] = wmma_bf16(af, bf0, acc[mt][0]);
          acc[mt][1] = wmma_bf16(af, bf1, acc[mt][1]);
        }
      }
      // epilogue: +bias (Q also *scale), store bf16 into per-wave LDS
#pragma unroll
      for (int mt = 0; mt < 4; ++mt)
#pragma unroll
        for (int nt = 0; nt < 2; ++nt) {
          const int n = nt * 16 + nlo;
          const float bias = b_qkv[cb + n];
#pragma unroll
          for (int r8 = 0; r8 < 8; ++r8) {
            const int m = mt * 16 + mhi + r8;
            const float v = acc[mt][nt][r8] + bias;
            if (t == 0)      sQ[m * HD + n] = (bf16_t)(v * qscale);
            else if (t == 1) sK[m * HD + n] = (bf16_t)v;
            else             sVT[n * NP + m] = (bf16_t)v;   // transposed
          }
        }
    }

    // ---- Phase C: S = Q * K^T  (K=32 -> single WMMA per 16x16 tile) --------
#pragma unroll
    for (int mt = 0; mt < 4; ++mt) {
      v16bf aq = frag_ld(sQ + mt * 16 * HD, HD, lane);
#pragma unroll
      for (int nt = 0; nt < 4; ++nt) {
        v16bf bk = frag_ld(sK + nt * 16 * HD, HD, lane);
        v8f s = vzero8();
        s = wmma_bf16(aq, bk, s);
#pragma unroll
        for (int r8 = 0; r8 < 8; ++r8)
          sS[(mt * 16 + mhi + r8) * NP + nt * 16 + nlo] = s[r8];
      }
    }

    // ---- Softmax + relative position bias; P (bf16) over dead Q/K ----------
#pragma unroll
    for (int rr = 0; rr < 2; ++rr) {
      const int m = lane * 2 + rr;                  // 64 rows over 32 lanes
      if (m < NTOK) {
        const int ym = m / WS7, xm = m % WS7;
        float mx = -1e30f;
        for (int n = 0; n < NTOK; ++n) {
          const int yn = n / WS7, xn = n % WS7;
          const int idx = (ym - yn + WS7 - 1) * (2 * WS7 - 1) + (xm - xn + WS7 - 1);
          const float s = sS[m * NP + n] + bias_table[idx * HEADS + h];
          sS[m * NP + n] = s;
          mx = fmaxf(mx, s);
        }
        float sum = 0.0f;
        for (int n = 0; n < NTOK; ++n) {
          const float e = __expf(sS[m * NP + n] - mx);
          sS[m * NP + n] = e;
          sum += e;
        }
        const float inv = 1.0f / sum;
        for (int n = 0; n < NP; ++n)
          sP[m * NP + n] = (n < NTOK) ? (bf16_t)(sS[m * NP + n] * inv)
                                      : (bf16_t)0.0f;
      } else {
        for (int n = 0; n < NP; ++n) sP[m * NP + n] = (bf16_t)0.0f;
      }
    }

    // ---- Phase D: O = P @ V   [64,64] x [64,32], kept in registers ---------
#pragma unroll
    for (int mt = 0; mt < 4; ++mt) {
      oacc[hIter][mt][0] = vzero8();
      oacc[hIter][mt][1] = vzero8();
    }
#pragma unroll
    for (int kk = 0; kk < 2; ++kk) {
      v16bf bv0 = frag_ld(sVT + 0 * 16 * NP + kk * 32, NP, lane);
      v16bf bv1 = frag_ld(sVT + 1 * 16 * NP + kk * 32, NP, lane);
#pragma unroll
      for (int mt = 0; mt < 4; ++mt) {
        v16bf ap = frag_ld(sP + mt * 16 * NP + kk * 32, NP, lane);
        oacc[hIter][mt][0] = wmma_bf16(ap, bv0, oacc[hIter][mt][0]);
        oacc[hIter][mt][1] = wmma_bf16(ap, bv1, oacc[hIter][mt][1]);
      }
    }
  } // hIter

  __syncthreads();   // all waves done with sX and attention

  // ---- Concatenate heads into out_heads (aliases dead sX region) -----------
  bf16_t* outH = (bf16_t*)smem;                      // [64][512]
#pragma unroll
  for (int hIter = 0; hIter < 2; ++hIter) {
    const int h = wave + 8 * hIter;
#pragma unroll
    for (int mt = 0; mt < 4; ++mt)
#pragma unroll
      for (int nt = 0; nt < 2; ++nt)
#pragma unroll
        for (int r8 = 0; r8 < 8; ++r8) {
          const int m = mt * 16 + mhi + r8;
          const int c = h * HD + nt * 16 + nlo;
          outH[m * CDIM + c] = (bf16_t)oacc[hIter][mt][nt][r8];
        }
  }
  __syncthreads();

  // ---- Phase E: projection [64,512] x [512,512] + bias -> global fp32 ------
  {
    v8f pacc[4][4];
#pragma unroll
    for (int mt = 0; mt < 4; ++mt)
#pragma unroll
      for (int ntl = 0; ntl < 4; ++ntl) pacc[mt][ntl] = vzero8();

    for (int kk = 0; kk < CDIM / 32; ++kk) {        // rolled K loop
      const int kb = kk * 32;
      v16bf af[4];
#pragma unroll
      for (int mt = 0; mt < 4; ++mt)
        af[mt] = frag_ld(outH + mt * 16 * CDIM + kb, CDIM, lane);
#pragma unroll
      for (int ntl = 0; ntl < 4; ++ntl) {
        const int j0 = (wave * 4 + ntl) * 16;
        v16bf bf = frag_ld(wprojT + (size_t)j0 * CDIM + kb, CDIM, lane);
#pragma unroll
        for (int mt = 0; mt < 4; ++mt)
          pacc[mt][ntl] = wmma_bf16(af[mt], bf, pacc[mt][ntl]);
      }
    }
#pragma unroll
    for (int mt = 0; mt < 4; ++mt)
#pragma unroll
      for (int ntl = 0; ntl < 4; ++ntl) {
        const int col  = (wave * 4 + ntl) * 16 + nlo;
        const float bias = b_proj[col];
#pragma unroll
        for (int r8 = 0; r8 < 8; ++r8) {
          const int m = mt * 16 + mhi + r8;
          if (m < NTOK)
            out[((size_t)b * NTOK + m) * CDIM + col] = pacc[mt][ntl][r8] + bias;
        }
      }
  }
}

// ---------------------------------------------------------------------------
extern "C" void kernel_launch(void* const* d_in, const int* in_sizes, int n_in,
                              void* d_out, int out_size, void* d_ws, size_t ws_size,
                              hipStream_t stream) {
  const float* x        = (const float*)d_in[0];
  const float* w_qkv    = (const float*)d_in[1];
  const float* b_qkv    = (const float*)d_in[2];
  const float* w_proj   = (const float*)d_in[3];
  const float* b_proj   = (const float*)d_in[4];
  const float* bias_tab = (const float*)d_in[5];
  // d_in[6]=num_heads(16), d_in[7]=window_size(7): baked in as constants.

  bf16_t* wqkvT  = (bf16_t*)d_ws;                               // 1536*512 bf16
  bf16_t* wprojT = (bf16_t*)((char*)d_ws + (size_t)1536 * 512 * 2);

  prep_weights<<<(3 * CDIM * CDIM + 255) / 256, 256, 0, stream>>>(
      w_qkv, w_proj, wqkvT, wprojT);

  win_attn_fused<<<4096, 256, SMEM_BYTES, stream>>>(
      x, wqkvT, b_qkv, wprojT, b_proj, bias_tab, (float*)d_out);
}
